// DoubleAttention_27693949125373
// MI455X (gfx1250) — matrix operand order repaired and checked by reading, not compiled, not run on hardware
//
#include <hip/hip_runtime.h>
#include <hip/hip_bf16.h>

typedef __bf16 bf16_t;
typedef __attribute__((ext_vector_type(8)))  bf16_t v8bf;
typedef __attribute__((ext_vector_type(16))) bf16_t v16bf;
typedef __attribute__((ext_vector_type(8)))  float  v8f;
typedef __attribute__((ext_vector_type(4)))  float  v4f;

union FragU { v16bf v; v8bf h[2]; };

__device__ __forceinline__ v16bf join16(v8bf lo, v8bf hi) {
  FragU u; u.h[0] = lo; u.h[1] = hi; return u.v;
}

// ---------------------------------------------------------------------------
// fp32 -> bf16 conversion (4 elements / thread)
// ---------------------------------------------------------------------------
__global__ __launch_bounds__(256)
void cvt_bf16_kernel(const float* __restrict__ src, bf16_t* __restrict__ dst, long count4) {
  long i = ((long)blockIdx.x * blockDim.x + threadIdx.x);
  if (i < count4) {
    v4f f = *(const v4f*)(src + i * 4);
    bf16_t* d = dst + i * 4;
    d[0] = (bf16_t)f.x; d[1] = (bf16_t)f.y; d[2] = (bf16_t)f.z; d[3] = (bf16_t)f.w;
  }
}

__global__ __launch_bounds__(256)
void bias_stack_kernel(const float* __restrict__ bk, const float* __restrict__ bq,
                       const float* __restrict__ bv, float* __restrict__ bqkv) {
  int i = blockIdx.x * blockDim.x + threadIdx.x;  // 0..1023
  float v = (i < 256) ? bk[i] : (i < 512) ? bq[i - 256] : bv[i - 512];
  if (i < 1024) bqkv[i] = v;
}

// ---------------------------------------------------------------------------
// Tiled bf16 WMMA GEMM:  C[M,N] = A[M,K] * B[K,N] (+bias[row]) (+residual)
// Block: 256 threads = 8 waves (2x4), wave tile 64x64, block tile 128x256, BK=32
// mode 0: write bf16 (Cb);  mode 1: write fp32 (Cf) with residual add
// ---------------------------------------------------------------------------
#define BM 128
#define BN 256
#define BK 32
#define LDS_PAD 40   // bf16 elements per padded row (80B -> conflict-free b128 reads)

__global__ __launch_bounds__(256)
void gemm_bf16_kernel(const bf16_t* __restrict__ A, int lda, long strideA,
                      const bf16_t* __restrict__ Bm, int ldb, long strideB,
                      int K,
                      const float* __restrict__ bias,
                      bf16_t* __restrict__ Cb, float* __restrict__ Cf,
                      const float* __restrict__ resid,
                      int ldc, long strideC, int mode) {
  __shared__ bf16_t As[BM][LDS_PAD];
  __shared__ bf16_t Bs[BN][LDS_PAD];

  const int tid  = threadIdx.x;
  const int lane = tid & 31;
  const int w    = tid >> 5;
  const int wm   = w >> 2;          // 0..1
  const int wn   = w & 3;           // 0..3
  const int m0   = blockIdx.y * BM;
  const int n0   = blockIdx.x * BN;
  const long z   = blockIdx.z;

  A  += z * strideA;
  Bm += z * strideB;

  const int nl  = lane & 15;
  const int hi  = lane >> 4;
  const int kbA = hi * 8;           // A K-half base
  const int kbB = hi * 16;          // unused; B fragments also use kbA

  v8f acc[4][4];
  #pragma unroll
  for (int i = 0; i < 4; ++i)
    #pragma unroll
    for (int j = 0; j < 4; ++j)
      acc[i][j] = (v8f)(0.0f);

  for (int k0 = 0; k0 < K; k0 += BK) {
    // stage A tile 128x32 (row-major), 2 chunks of 8 bf16 per thread
    #pragma unroll
    for (int c = 0; c < 2; ++c) {
      int chunk = tid + 256 * c;            // 0..511
      int r  = chunk >> 2;                  // 0..127
      int kc = (chunk & 3) * 8;             // 0,8,16,24
      v8bf va = *(const v8bf*)(A + (long)(m0 + r) * lda + (k0 + kc));
      *(v8bf*)&As[r][kc] = va;
    }
    // stage B tile 32x256 transposed -> Bs[n][k]
    #pragma unroll
    for (int c = 0; c < 4; ++c) {
      int chunk = tid + 256 * c;            // 0..1023
      int kr = chunk >> 5;                  // 0..31
      int nc = (chunk & 31) * 8;            // 0..248
      v8bf vb = *(const v8bf*)(Bm + (long)(k0 + kr) * ldb + (n0 + nc));
      #pragma unroll
      for (int jj = 0; jj < 8; ++jj) Bs[nc + jj][kr] = vb[jj];
    }
    __syncthreads();

    v16bf af[4], bfr[4];
    #pragma unroll
    for (int i = 0; i < 4; ++i) {
      int r = wm * 64 + i * 16 + nl;
      af[i] = join16(*(const v8bf*)&As[r][kbA], *(const v8bf*)&As[r][kbA + 16]);
    }
    #pragma unroll
    for (int j = 0; j < 4; ++j) {
      int cidx = wn * 64 + j * 16 + nl;
      bfr[j] = join16(*(const v8bf*)&Bs[cidx][kbA], *(const v8bf*)&Bs[cidx][kbA + 16]);
    }
    #pragma unroll
    for (int i = 0; i < 4; ++i)
      #pragma unroll
      for (int j = 0; j < 4; ++j)
        acc[i][j] = __builtin_amdgcn_wmma_f32_16x16x32_bf16(
            false, af[i], false, bfr[j], (short)0, acc[i][j], false, false);
    __syncthreads();
  }

  // epilogue
  if (mode == 0) {
    bf16_t* C = Cb + z * strideC;
    #pragma unroll
    for (int i = 0; i < 4; ++i)
      #pragma unroll
      for (int j = 0; j < 4; ++j) {
        int col = n0 + wn * 64 + j * 16 + nl;
        #pragma unroll
        for (int e = 0; e < 8; ++e) {
          int row = m0 + wm * 64 + i * 16 + e + hi * 8;
          C[(long)row * ldc + col] = (bf16_t)(acc[i][j][e] + bias[row]);
        }
      }
  } else {
    float* C = Cf + z * strideC;
    const float* R = resid + z * strideC;
    #pragma unroll
    for (int i = 0; i < 4; ++i)
      #pragma unroll
      for (int j = 0; j < 4; ++j) {
        int col = n0 + wn * 64 + j * 16 + nl;
        #pragma unroll
        for (int e = 0; e < 8; ++e) {
          int row = m0 + wm * 64 + i * 16 + e + hi * 8;
          long idx = (long)row * ldc + col;
          C[idx] = acc[i][j][e] + bias[row] + R[idx];
        }
      }
  }
}

// ---------------------------------------------------------------------------
// Key softmax over L=4096 (in-place on P rows 0..255 per batch). Block = 1 row.
// ---------------------------------------------------------------------------
__global__ __launch_bounds__(256)
void softmax_row_kernel(bf16_t* __restrict__ P) {
  bf16_t* row = P + (long)blockIdx.y * (1024L * 4096) + (long)blockIdx.x * 4096;
  const int tid = threadIdx.x;
  __shared__ float red[256];
  float v[16];
  float m = -1e30f;
  #pragma unroll
  for (int i = 0; i < 16; ++i) { v[i] = (float)row[tid + i * 256]; m = fmaxf(m, v[i]); }
  red[tid] = m; __syncthreads();
  for (int s = 128; s > 0; s >>= 1) { if (tid < s) red[tid] = fmaxf(red[tid], red[tid + s]); __syncthreads(); }
  m = red[0]; __syncthreads();
  float sum = 0.0f;
  #pragma unroll
  for (int i = 0; i < 16; ++i) { v[i] = __expf(v[i] - m); sum += v[i]; }
  red[tid] = sum; __syncthreads();
  for (int s = 128; s > 0; s >>= 1) { if (tid < s) red[tid] += red[tid + s]; __syncthreads(); }
  float inv = 1.0f / red[0];
  #pragma unroll
  for (int i = 0; i < 16; ++i) row[tid + i * 256] = (bf16_t)(v[i] * inv);
}

// ---------------------------------------------------------------------------
// Query softmax over 32 head-channels (in-place on P rows 256..511).
// grid: (L/256, heads, batch); one thread per spatial position.
// ---------------------------------------------------------------------------
__global__ __launch_bounds__(256)
void softmax_chan_kernel(bf16_t* __restrict__ P) {
  int l = blockIdx.x * 256 + threadIdx.x;
  bf16_t* base = P + (long)blockIdx.z * (1024L * 4096)
                   + (long)(256 + blockIdx.y * 32) * 4096 + l;
  float v[32];
  float m = -1e30f;
  #pragma unroll
  for (int k = 0; k < 32; ++k) { v[k] = (float)base[(long)k * 4096]; m = fmaxf(m, v[k]); }
  float sum = 0.0f;
  #pragma unroll
  for (int k = 0; k < 32; ++k) { v[k] = __expf(v[k] - m); sum += v[k]; }
  float inv = 1.0f / sum;
  #pragma unroll
  for (int k = 0; k < 32; ++k) base[(long)k * 4096] = (bf16_t)(v[k] * inv);
}

// ---------------------------------------------------------------------------
// Context: ctx[n,h,v,k] = sum_l V[v,l]*Ksm[k,l]   (64x32, K=4096 via WMMA)
// 8 waves split K; deterministic LDS partial-sum reduction (no atomics).
// A and B fragments load straight from global (both contiguous 16B runs).
// ---------------------------------------------------------------------------
__global__ __launch_bounds__(256)
void context_kernel(const bf16_t* __restrict__ P, float* __restrict__ ctx) {
  const int h = blockIdx.x, n = blockIdx.y;
  const bf16_t* Pb = P + (long)n * (1024L * 4096);
  const bf16_t* V  = Pb + (long)(512 + h * 64) * 4096;
  const bf16_t* Ks = Pb + (long)(h * 32) * 4096;

  const int tid = threadIdx.x, lane = tid & 31, w = tid >> 5;
  const int nl = lane & 15, hi = lane >> 4;

  v8f acc[4][2];
  #pragma unroll
  for (int i = 0; i < 4; ++i) { acc[i][0] = (v8f)(0.0f); acc[i][1] = (v8f)(0.0f); }

  const int l0 = w * 512;
  for (int lt = 0; lt < 16; ++lt) {
    int l = l0 + lt * 32;
    v16bf af[4], bfr[2];
    #pragma unroll
    for (int i = 0; i < 4; ++i) {
      const bf16_t* p = V + (long)(i * 16 + nl) * 4096 + l + hi * 8;
      af[i] = join16(*(const v8bf*)p, *(const v8bf*)(p + 16));
    }
    #pragma unroll
    for (int j = 0; j < 2; ++j) {
      const bf16_t* p = Ks + (long)(j * 16 + nl) * 4096 + l + hi * 8;
      bfr[j] = join16(*(const v8bf*)p, *(const v8bf*)(p + 16));
    }
    #pragma unroll
    for (int i = 0; i < 4; ++i)
      #pragma unroll
      for (int j = 0; j < 2; ++j)
        acc[i][j] = __builtin_amdgcn_wmma_f32_16x16x32_bf16(
            false, af[i], false, bfr[j], (short)0, acc[i][j], false, false);
  }

  __shared__ float part[8][64][32];   // 64 KB of the WGP's 320 KB LDS
  #pragma unroll
  for (int i = 0; i < 4; ++i)
    #pragma unroll
    for (int j = 0; j < 2; ++j)
      #pragma unroll
      for (int e = 0; e < 8; ++e)
        part[w][i * 16 + e + hi * 8][j * 16 + nl] = acc[i][j][e];
  __syncthreads();

  float* out = ctx + ((long)n * 8 + h) * 2048;
  for (int t = tid; t < 2048; t += 256) {
    float s = 0.0f;
    #pragma unroll
    for (int w2 = 0; w2 < 8; ++w2) s += part[w2][t >> 5][t & 31];
    out[t] = s;
  }
}

// ---------------------------------------------------------------------------
// Fold We: Mst[n, c, h*32+k] = sum_v We[c, h*64+v] * ctx[n,h,v,k]  (fp32 VALU)
// ---------------------------------------------------------------------------
__global__ __launch_bounds__(256)
void mix_kernel(const float* __restrict__ We, const float* __restrict__ ctx,
                bf16_t* __restrict__ Mst) {
  const int h = blockIdx.x, n = blockIdx.y;
  __shared__ float cs[64][32];
  const float* c = ctx + ((long)n * 8 + h) * 2048;
  for (int t = threadIdx.x; t < 2048; t += 256) cs[t >> 5][t & 31] = c[t];
  __syncthreads();
  const int kc = threadIdx.x & 31;
  const int g  = threadIdx.x >> 5;            // 0..7
  for (int ci = 0; ci < 64; ++ci) {
    int cc = g + ci * 8;                      // 0..511
    const float* wrow = We + (long)cc * 512 + h * 64;
    float s = 0.0f;
    #pragma unroll
    for (int v = 0; v < 64; ++v) s += wrow[v] * cs[v][kc];
    Mst[((long)n * 512 + cc) * 256 + h * 32 + kc] = (bf16_t)s;
  }
}

// ---------------------------------------------------------------------------
extern "C" void kernel_launch(void* const* d_in, const int* in_sizes, int n_in,
                              void* d_out, int out_size, void* d_ws, size_t ws_size,
                              hipStream_t stream) {
  const float* x  = (const float*)d_in[0];
  const float* Wk = (const float*)d_in[1];
  const float* bk = (const float*)d_in[2];
  const float* Wq = (const float*)d_in[3];
  const float* bq = (const float*)d_in[4];
  const float* Wv = (const float*)d_in[5];
  const float* bv = (const float*)d_in[6];
  const float* We = (const float*)d_in[7];
  const float* be = (const float*)d_in[8];
  float* out = (float*)d_out;

  const long N = 16, C = 512, L = 4096;

  // workspace layout
  char* ws = (char*)d_ws;
  size_t off = 0;
  auto take = [&](size_t bytes) { char* p = ws + off; off += (bytes + 255) & ~(size_t)255; return p; };
  bf16_t* Xb    = (bf16_t*)take(N * C * L * 2);          // 67 MB
  bf16_t* Wqkvb = (bf16_t*)take(1024L * 512 * 2);        // 1 MB
  float*  bqkv  = (float*) take(1024L * 4);
  bf16_t* P     = (bf16_t*)take(N * 1024L * L * 2);      // 134 MB (K|Q|V stacked)
  float*  ctx   = (float*) take(N * 8L * 64 * 32 * 4);   // 1 MB
  bf16_t* Mst   = (bf16_t*)take(N * 512L * 256 * 2);     // 4 MB
  (void)ws_size;

  // 1) conversions
  {
    long c4 = (N * C * L) / 4;
    cvt_bf16_kernel<<<dim3((unsigned)(c4 / 256)), 256, 0, stream>>>(x, Xb, c4);
    cvt_bf16_kernel<<<dim3(128), 256, 0, stream>>>(Wk, Wqkvb,                131072 / 4);
    cvt_bf16_kernel<<<dim3(128), 256, 0, stream>>>(Wq, Wqkvb + 131072,       131072 / 4);
    cvt_bf16_kernel<<<dim3(256), 256, 0, stream>>>(Wv, Wqkvb + 262144,       262144 / 4);
    bias_stack_kernel<<<dim3(4), 256, 0, stream>>>(bk, bq, bv, bqkv);
  }

  // 2) QKV GEMM: P[n,1024,L] = Wqkv(1024x512) * Xb[n](512xL) + bqkv  (bf16 out)
  gemm_bf16_kernel<<<dim3(16, 8, 16), 256, 0, stream>>>(
      Wqkvb, 512, 0L,
      Xb, 4096, C * L,
      512, bqkv,
      P, nullptr, nullptr,
      4096, 1024L * L, 0);

  // 3) softmaxes (in-place on P)
  softmax_row_kernel <<<dim3(256, 16),    256, 0, stream>>>(P);
  softmax_chan_kernel<<<dim3(16, 8, 16),  256, 0, stream>>>(P);

  // 4) per-head context GEMMs (WMMA, K=4096 split over 8 waves)
  context_kernel<<<dim3(8, 16), 256, 0, stream>>>(P, ctx);

  // 5) fold We into context
  mix_kernel<<<dim3(8, 16), 256, 0, stream>>>(We, ctx, Mst);

  // 6) final GEMM: out[n] = Mst[n](512x256) * Qsm[n](256xL) + be + x   (fp32 out)
  gemm_bf16_kernel<<<dim3(16, 4, 16), 256, 0, stream>>>(
      Mst, 256, 512L * 256,
      P + 256L * 4096, 4096, 1024L * L,
      256, be,
      nullptr, out, x,
      4096, 512L * L, 1);
}
